// MambaBlock_57836029608262
// MI455X (gfx1250) — matrix operand order repaired and checked by reading, not compiled, not run on hardware
//
#include <hip/hip_runtime.h>
#include <hip/hip_bf16.h>

// ---------------------------------------------------------------------------
// Bidirectional Mamba block for MI455X (gfx1250, wave32).
// GEMMs: bf16 operands, f32 accumulation via v_wmma_f32_16x16x32_bf16,
//        2x2 register-blocked (32x32 output per wave -> 2 b128 loads / wmma),
//        scalar tile indices (readfirstlane) so the K-loop has no EXEC ops.
// Scan / conv / activations: f32 VALU (v_exp_f32).
// Workspace use: ~273 MB of d_ws.
// ---------------------------------------------------------------------------

#define DD_IN    512
#define DD_MODEL 256
#define DD_STATE 128
#define DD_CONV  4
#define DD_INNER 512
#define DDT_RANK 16
#define NB       8
#define NL       2048
#define MTOK     (NB * NL)   // 16384 tokens

typedef __attribute__((ext_vector_type(16))) __bf16 v16bf;
typedef __attribute__((ext_vector_type(8)))  float  v8f;

union FragBF { v16bf v; uint4 q[2]; };

__device__ __forceinline__ unsigned short f2bf(float f) {
    unsigned int u = __float_as_uint(f);
    unsigned int r = (u + 0x7FFFu + ((u >> 16) & 1u)) >> 16;   // RNE
    return (unsigned short)r;
}
__device__ __forceinline__ float dsilu(float v) { return v / (1.f + __expf(-v)); }

// --------------------------- elementwise helpers ---------------------------

__global__ __launch_bounds__(256) void k_f32_to_bf16(const float* __restrict__ s,
                                                     unsigned short* __restrict__ d,
                                                     long n) {
    long i = (long)blockIdx.x * 256 + threadIdx.x;
    if (i < n) d[i] = f2bf(s[i]);
}

// zero-padded strided f32 -> bf16 repack (rows x cols -> rows x drow)
__global__ __launch_bounds__(256) void k_pack_pad(const float* __restrict__ src, int srow,
                                                  int cols, unsigned short* __restrict__ dst,
                                                  int drow, long total) {
    long i = (long)blockIdx.x * 256 + threadIdx.x;
    if (i >= total) return;
    int  c = (int)(i % drow);
    long r = i / drow;
    dst[i] = (c < cols) ? f2bf(src[r * (long)srow + c]) : (unsigned short)0;
}

// per-(b, feature) L2-norm over the sequence axis
__global__ __launch_bounds__(256) void k_norm(const float* __restrict__ x,
                                              float* __restrict__ inv) {
    __shared__ float red[256];
    int bd = blockIdx.x;              // 0 .. NB*512-1
    int b  = bd >> 9, dch = bd & 511;
    const float* p = x + ((size_t)b * NL) * DD_IN + dch;
    float s = 0.f;
    for (int l = threadIdx.x; l < NL; l += 256) { float v = p[(size_t)l * DD_IN]; s += v * v; }
    red[threadIdx.x] = s; __syncthreads();
    for (int w = 128; w > 0; w >>= 1) {
        if (threadIdx.x < w) red[threadIdx.x] += red[threadIdx.x + w];
        __syncthreads();
    }
    if (threadIdx.x == 0) inv[bd] = 1.f / fmaxf(sqrtf(red[0]), 1e-12f);
}

__global__ __launch_bounds__(256) void k_xnorm(const float* __restrict__ x,
                                               const float* __restrict__ inv,
                                               unsigned short* __restrict__ xn) {
    long i = (long)blockIdx.x * 256 + threadIdx.x;
    if (i >= (long)MTOK * DD_IN) return;
    int  d = (int)(i & 511);
    long m = i >> 9;
    int  b = (int)(m >> 11);
    xn[i] = f2bf(x[i] * inv[(b << 9) + d]);
}

// causal (dir=0) / anti-causal (dir=1) depthwise conv + SiLU
__global__ __launch_bounds__(256) void k_conv(const float* __restrict__ xz,
                                              const float* __restrict__ cw,
                                              const float* __restrict__ cb,
                                              float* __restrict__ uo,
                                              unsigned short* __restrict__ uob, int dir) {
    long i = (long)blockIdx.x * 256 + threadIdx.x;
    if (i >= (long)MTOK * DD_INNER) return;
    int  d = (int)(i & 511);
    long m = i >> 9;
    int  b = (int)(m >> 11), l = (int)(m & 2047);
    float acc = cb[d];
#pragma unroll
    for (int k = 0; k < DD_CONV; ++k) {
        int lk = dir ? (l + 3 - k) : (l - 3 + k);
        if (lk >= 0 && lk < NL)
            acc += cw[d * 4 + k] * xz[(((long)b << 11) + lk) * 1024 + d];
    }
    float s = dsilu(acc);
    uo[i]  = s;
    uob[i] = f2bf(s);
}

// fusion: (x_f + x_b) * z   -> bf16 for the final GEMM
__global__ __launch_bounds__(256) void k_fuse(const float* __restrict__ fus,
                                              const float* __restrict__ z,
                                              unsigned short* __restrict__ out) {
    long i = (long)blockIdx.x * 256 + threadIdx.x;
    if (i < (long)MTOK * DD_MODEL) out[i] = f2bf(fus[i] * z[i]);
}

// ------------------------------- WMMA GEMM ---------------------------------
// C(M,N) = act( A(M,K)bf16 @ W(N,K)bf16^T + bias ) (+resid)
// 2x2 register blocking: one wave owns a 32x32 output tile (4 accumulators),
// so each A/B fragment feeds two WMMAs -> 2 b128 loads per wmma.
// Tile indices are forced into SGPRs (readfirstlane) so the odd-N-tile case
// is a scalar-branch loop specialization, not per-iteration EXEC masking.
// act: 0=none 1=leaky(0.01) 2=silu 3=softplus
__global__ __launch_bounds__(256) void k_gemm(const unsigned short* __restrict__ A,
                                              const unsigned short* __restrict__ W,
                                              const float* __restrict__ bias,
                                              const float* __restrict__ resid,
                                              float* __restrict__ outf,
                                              unsigned short* __restrict__ outb,
                                              int M, int N, int K, int act, int accum) {
    int wave = __builtin_amdgcn_readfirstlane((int)((blockIdx.x * 256 + threadIdx.x) >> 5));
    int lane = threadIdx.x & 31;
    int tilesN  = N >> 4;
    int tilesN2 = (tilesN + 1) >> 1;
    int tm2 = wave / tilesN2, tn2 = wave - tm2 * tilesN2;
    if (tm2 >= (M >> 5)) return;                // scalar exit, EXEC stays full
    int  h = lane >> 4, r16 = lane & 15;
    bool hasN1 = (tn2 * 2 + 1) < tilesN;        // scalar (N=272 -> odd tile count)

    const unsigned short* arow0 = A + (size_t)(tm2 * 32 + r16) * K;
    const unsigned short* arow1 = arow0 + (size_t)16 * K;
    const unsigned short* wrow0 = W + (size_t)(tn2 * 32 + r16) * K;
    const unsigned short* wrow1 = wrow0 + (size_t)16 * K;

    v8f acc00 = {0.f, 0.f, 0.f, 0.f, 0.f, 0.f, 0.f, 0.f};
    v8f acc01 = acc00, acc10 = acc00, acc11 = acc00;

    if (hasN1) {
        for (int k0 = 0; k0 < K; k0 += 32) {
            FragBF fa0, fa1, fb0, fb1;
            // A fragment (16-bit A 16x32 layout): e<8 -> k0+8h+e ; e>=8 -> k0+16+8h+(e-8)
            fa0.q[0] = *(const uint4*)(arow0 + k0 + 8 * h);
            fa0.q[1] = *(const uint4*)(arow0 + k0 + 16 + 8 * h);
            fa1.q[0] = *(const uint4*)(arow1 + k0 + 8 * h);
            fa1.q[1] = *(const uint4*)(arow1 + k0 + 16 + 8 * h);
            // B fragment (16-bit B 32x16 layout): lane=n, e<8 -> k0+8h+e ; e>=8 -> k0+16+8h+(e-8)
            fb0.q[0] = *(const uint4*)(wrow0 + k0 + 8 * h);
            fb0.q[1] = *(const uint4*)(wrow0 + k0 + 16 + 8 * h);
            fb1.q[0] = *(const uint4*)(wrow1 + k0 + 8 * h);
            fb1.q[1] = *(const uint4*)(wrow1 + k0 + 16 + 8 * h);
            acc00 = __builtin_amdgcn_wmma_f32_16x16x32_bf16(false, fa0.v, false, fb0.v,
                                                            (short)0, acc00, false, false);
            acc10 = __builtin_amdgcn_wmma_f32_16x16x32_bf16(false, fa1.v, false, fb0.v,
                                                            (short)0, acc10, false, false);
            acc01 = __builtin_amdgcn_wmma_f32_16x16x32_bf16(false, fa0.v, false, fb1.v,
                                                            (short)0, acc01, false, false);
            acc11 = __builtin_amdgcn_wmma_f32_16x16x32_bf16(false, fa1.v, false, fb1.v,
                                                            (short)0, acc11, false, false);
        }
    } else {
        for (int k0 = 0; k0 < K; k0 += 32) {
            FragBF fa0, fa1, fb0;
            fa0.q[0] = *(const uint4*)(arow0 + k0 + 8 * h);
            fa0.q[1] = *(const uint4*)(arow0 + k0 + 16 + 8 * h);
            fa1.q[0] = *(const uint4*)(arow1 + k0 + 8 * h);
            fa1.q[1] = *(const uint4*)(arow1 + k0 + 16 + 8 * h);
            fb0.q[0] = *(const uint4*)(wrow0 + k0 + 8 * h);
            fb0.q[1] = *(const uint4*)(wrow0 + k0 + 16 + 8 * h);
            acc00 = __builtin_amdgcn_wmma_f32_16x16x32_bf16(false, fa0.v, false, fb0.v,
                                                            (short)0, acc00, false, false);
            acc10 = __builtin_amdgcn_wmma_f32_16x16x32_bf16(false, fa1.v, false, fb0.v,
                                                            (short)0, acc10, false, false);
        }
    }

    auto emit = [&](const v8f& acc, int tmi, int tni) {
        int n = tni * 16 + r16;
        float bv = bias ? bias[n] : 0.f;
#pragma unroll
        for (int r = 0; r < 8; ++r) {
            int m = tmi * 16 + 8 * h + r;
            float v = acc[r] + bv;
            if (act == 1)      v = v > 0.f ? v : 0.01f * v;
            else if (act == 2) v = dsilu(v);
            else if (act == 3) v = (v > 20.f) ? v : log1pf(__expf(v));
            size_t o = (size_t)m * N + n;
            if (resid) v += resid[o];
            if (outf)  outf[o] = accum ? (outf[o] + v) : v;
            if (outb)  outb[o] = f2bf(v);
        }
    };
    emit(acc00, tm2 * 2,     tn2 * 2);
    emit(acc10, tm2 * 2 + 1, tn2 * 2);
    if (hasN1) {
        emit(acc01, tm2 * 2,     tn2 * 2 + 1);
        emit(acc11, tm2 * 2 + 1, tn2 * 2 + 1);
    }
}

// ------------------------------ selective scan -----------------------------
// One wave per (b, d_inner) row; 4 state elems / lane (4 x 32 = 128).
__global__ __launch_bounds__(256) void k_scan(const float* __restrict__ delta,
                                              const float* __restrict__ dbl,
                                              const float* __restrict__ uconv,
                                              const float* __restrict__ xz,
                                              const float* __restrict__ Alog,
                                              const float* __restrict__ Dp,
                                              unsigned short* __restrict__ ybf, int dir) {
    int wave = (blockIdx.x * 256 + threadIdx.x) >> 5;
    int lane = threadIdx.x & 31;
    if (wave >= NB * DD_INNER) return;
    int b = wave >> 9, d = wave & 511;

    const float LOG2E = 1.44269504088896340736f;
    float4 al = *(const float4*)(Alog + (size_t)d * DD_STATE + lane * 4);
    float4 aL;  // -exp(Alog) * log2(e)  (use exp2 in the hot loop)
    aL.x = -__expf(al.x) * LOG2E; aL.y = -__expf(al.y) * LOG2E;
    aL.z = -__expf(al.z) * LOG2E; aL.w = -__expf(al.w) * LOG2E;
    float Dd = Dp[d];

    float4 h4 = {0.f, 0.f, 0.f, 0.f};
    for (int t = 0; t < NL; ++t) {
        int  tt = dir ? (NL - 1 - t) : t;
        long m  = ((long)b << 11) + tt;

        float dlt = delta[m * DD_INNER + d];
        float uu  = uconv[m * DD_INNER + d];
        float zz  = xz[m * 1024 + DD_INNER + d];
        float4 b4 = *(const float4*)(dbl + m * 272 + DDT_RANK + lane * 4);
        float4 c4 = *(const float4*)(dbl + m * 272 + DDT_RANK + DD_STATE + lane * 4);

        if (t + 1 < NL) {   // hint next timestep's rows into cache
            long mn = ((long)b << 11) + (dir ? (NL - 2 - t) : (t + 1));
            __builtin_prefetch(dbl + mn * 272 + DDT_RANK + lane * 4, 0, 1);
            __builtin_prefetch(delta + mn * DD_INNER + d, 0, 1);
        }

        float du = dlt * uu;
        h4.x = __builtin_amdgcn_exp2f(dlt * aL.x) * h4.x + du * b4.x;
        h4.y = __builtin_amdgcn_exp2f(dlt * aL.y) * h4.y + du * b4.y;
        h4.z = __builtin_amdgcn_exp2f(dlt * aL.z) * h4.z + du * b4.z;
        h4.w = __builtin_amdgcn_exp2f(dlt * aL.w) * h4.w + du * b4.w;

        float p = h4.x * c4.x + h4.y * c4.y + h4.z * c4.z + h4.w * c4.w;
        p += __shfl_xor(p, 16, 32);
        p += __shfl_xor(p, 8, 32);
        p += __shfl_xor(p, 4, 32);
        p += __shfl_xor(p, 2, 32);
        p += __shfl_xor(p, 1, 32);

        if (lane == 0) {
            float y = (p + Dd * uu) * dsilu(zz);
            ybf[m * DD_INNER + d] = f2bf(y);
        }
    }
}

// --------------------------------- host ------------------------------------

static inline int cdiv(long a, long b) { return (int)((a + b - 1) / b); }

static void launch_gemm(hipStream_t s, const unsigned short* A, const unsigned short* W,
                        const float* bias, const float* resid, float* outf,
                        unsigned short* outb, int M, int N, int K, int act, int accum) {
    long tilesN2 = ((N / 16) + 1) / 2;
    long waves   = (long)(M / 32) * tilesN2;
    hipLaunchKernelGGL(k_gemm, dim3(cdiv(waves * 32, 256)), dim3(256), 0, s,
                       A, W, bias, resid, outf, outb, M, N, K, act, accum);
}

extern "C" void kernel_launch(void* const* d_in, const int* in_sizes, int n_in,
                              void* d_out, int out_size, void* d_ws, size_t ws_size,
                              hipStream_t stream) {
    const float* x        = (const float*)d_in[0];
    const float* mlp_w    = (const float*)d_in[1];
    const float* mlp_b    = (const float*)d_in[2];
    const float* mlpres_w = (const float*)d_in[3];
    const float* mlpres_b = (const float*)d_in[4];
    const float* fl_w     = (const float*)d_in[5];
    const float* fl_b     = (const float*)d_in[6];
    const float* bl_w     = (const float*)d_in[7];
    const float* bl_b     = (const float*)d_in[8];
    const float* out_w    = (const float*)d_in[9];
    const float* out_b    = (const float*)d_in[10];
    // per-direction params: base 11 (fwd) and 20 (bwd)
    // [0]=in_w [1]=conv_w [2]=conv_b [3]=xproj_w [4]=dt_w [5]=dt_b [6]=Alog [7]=D [8]=outp_w

    char*  ws  = (char*)d_ws;
    size_t off = 0;
    auto alloc = [&](size_t bytes) -> void* {
        void* p = ws + off;
        off = (off + bytes + 255) & ~(size_t)255;
        return p;
    };

    const long Mt = MTOK;
    unsigned short* xn_bf   = (unsigned short*)alloc((size_t)Mt * DD_IN * 2);
    unsigned short* xup_bf  = (unsigned short*)alloc((size_t)Mt * DD_MODEL * 2);
    unsigned short* xfin_bf = (unsigned short*)alloc((size_t)Mt * DD_MODEL * 2);
    unsigned short* xbin_bf = (unsigned short*)alloc((size_t)Mt * DD_MODEL * 2);
    float*          z_f32   = (float*)alloc((size_t)Mt * DD_MODEL * 4);
    float*          xz_f32  = (float*)alloc((size_t)Mt * 1024 * 4);
    float*          uconv   = (float*)alloc((size_t)Mt * DD_INNER * 4);
    unsigned short* uconv_bf= (unsigned short*)alloc((size_t)Mt * DD_INNER * 2);
    float*          dbl     = (float*)alloc((size_t)Mt * 272 * 4);
    unsigned short* dlow_bf = (unsigned short*)alloc((size_t)Mt * 32 * 2);
    float*          delta   = (float*)alloc((size_t)Mt * DD_INNER * 4);
    unsigned short* y_bf    = (unsigned short*)alloc((size_t)Mt * DD_INNER * 2);
    float*          fus_f32 = (float*)alloc((size_t)Mt * DD_MODEL * 4);
    unsigned short* fus_bf  = (unsigned short*)alloc((size_t)Mt * DD_MODEL * 2);
    float*          inv_nrm = (float*)alloc((size_t)NB * DD_IN * 4);
    unsigned short* mlpw_bf   = (unsigned short*)alloc(256 * 512 * 2);
    unsigned short* mlpresw_bf= (unsigned short*)alloc(256 * 512 * 2);
    unsigned short* flw_bf    = (unsigned short*)alloc(256 * 256 * 2);
    unsigned short* blw_bf    = (unsigned short*)alloc(256 * 256 * 2);
    unsigned short* outw_bf   = (unsigned short*)alloc(512 * 256 * 2);
    unsigned short* inw_bf    = (unsigned short*)alloc(1024 * 256 * 2);
    unsigned short* xprojw_bf = (unsigned short*)alloc(272 * 512 * 2);
    unsigned short* dtw_bf    = (unsigned short*)alloc(512 * 32 * 2);
    unsigned short* outpw_bf  = (unsigned short*)alloc(256 * 512 * 2);
    (void)ws_size; (void)n_in; (void)in_sizes; (void)out_size;

    auto cvt = [&](const float* s, unsigned short* d, long n) {
        hipLaunchKernelGGL(k_f32_to_bf16, dim3(cdiv(n, 256)), dim3(256), 0, stream, s, d, n);
    };

    // stage shared weights
    cvt(mlp_w, mlpw_bf, 256 * 512);
    cvt(mlpres_w, mlpresw_bf, 256 * 512);
    cvt(fl_w, flw_bf, 256 * 256);
    cvt(bl_w, blw_bf, 256 * 256);
    cvt(out_w, outw_bf, 512 * 256);

    // norm + bf16 staging of x_norm
    hipLaunchKernelGGL(k_norm, dim3(NB * DD_IN), dim3(256), 0, stream, x, inv_nrm);
    hipLaunchKernelGGL(k_xnorm, dim3(cdiv((long)Mt * DD_IN, 256)), dim3(256), 0, stream,
                       x, inv_nrm, xn_bf);

    // top-level projections
    launch_gemm(stream, xn_bf, mlpw_bf, mlp_b, nullptr, nullptr, xup_bf,
                Mt, DD_MODEL, DD_IN, 0, 0);                                   // x_up
    launch_gemm(stream, xn_bf, mlpresw_bf, mlpres_b, nullptr, z_f32, nullptr,
                Mt, DD_MODEL, DD_IN, 1, 0);                                   // z (leaky)
    launch_gemm(stream, xup_bf, flw_bf, fl_b, nullptr, nullptr, xfin_bf,
                Mt, DD_MODEL, DD_MODEL, 0, 0);                                // fwd input
    launch_gemm(stream, xup_bf, blw_bf, bl_b, nullptr, nullptr, xbin_bf,
                Mt, DD_MODEL, DD_MODEL, 0, 0);                                // bwd input

    for (int dir = 0; dir < 2; ++dir) {
        int base = 11 + dir * 9;
        const float* in_w    = (const float*)d_in[base + 0];
        const float* conv_w  = (const float*)d_in[base + 1];
        const float* conv_b  = (const float*)d_in[base + 2];
        const float* xproj_w = (const float*)d_in[base + 3];
        const float* dt_w    = (const float*)d_in[base + 4];
        const float* dt_b    = (const float*)d_in[base + 5];
        const float* Alog    = (const float*)d_in[base + 6];
        const float* Dp      = (const float*)d_in[base + 7];
        const float* outp_w  = (const float*)d_in[base + 8];

        cvt(in_w, inw_bf, 1024 * 256);
        cvt(xproj_w, xprojw_bf, 272 * 512);
        cvt(outp_w, outpw_bf, 256 * 512);
        hipLaunchKernelGGL(k_pack_pad, dim3(cdiv(512L * 32, 256)), dim3(256), 0, stream,
                           dt_w, 16, 16, dtw_bf, 32, 512L * 32);              // dt_w pad K->32

        // xz = in @ in_w^T   (flip is a no-op for pointwise GEMMs)
        launch_gemm(stream, dir ? xbin_bf : xfin_bf, inw_bf, nullptr, nullptr,
                    xz_f32, nullptr, Mt, 1024, DD_MODEL, 0, 0);

        // depthwise conv (+SiLU), direction-aware
        hipLaunchKernelGGL(k_conv, dim3(cdiv((long)Mt * DD_INNER, 256)), dim3(256), 0, stream,
                           xz_f32, conv_w, conv_b, uconv, uconv_bf, dir);

        // dbl = u_conv @ xproj_w^T
        launch_gemm(stream, uconv_bf, xprojw_bf, nullptr, nullptr, dbl, nullptr,
                    Mt, 272, DD_INNER, 0, 0);

        // delta = softplus(dbl[:, :16] @ dt_w^T + dt_b)   (K padded 16 -> 32)
        hipLaunchKernelGGL(k_pack_pad, dim3(cdiv((long)Mt * 32, 256)), dim3(256), 0, stream,
                           dbl, 272, 16, dlow_bf, 32, (long)Mt * 32);
        launch_gemm(stream, dlow_bf, dtw_bf, dt_b, nullptr, delta, nullptr,
                    Mt, DD_INNER, 32, 3, 0);

        // selective scan (direction-aware), fused D*u and SiLU(z) gate -> y_bf
        hipLaunchKernelGGL(k_scan, dim3((NB * DD_INNER) / 8), dim3(256), 0, stream,
                           delta, dbl, uconv, xz_f32, Alog, Dp, y_bf, dir);

        // per-direction out-projection; accumulate fwd+bwd into fus_f32
        launch_gemm(stream, y_bf, outpw_bf, nullptr, nullptr, fus_f32, nullptr,
                    Mt, DD_MODEL, DD_INNER, 0, dir);
    }

    // (x_f + x_b) * z  -> bf16
    hipLaunchKernelGGL(k_fuse, dim3(cdiv((long)Mt * DD_MODEL, 256)), dim3(256), 0, stream,
                       fus_f32, z_f32, fus_bf);

    // final: fusion @ out_w^T + out_b + x
    launch_gemm(stream, fus_bf, outw_bf, out_b, x, (float*)d_out, nullptr,
                Mt, DD_IN, DD_MODEL, 0, 0);
}
